// inst_encoder_549755814581
// MI455X (gfx1250) — matrix operand order repaired and checked by reading, not compiled, not run on hardware
//
#include <hip/hip_runtime.h>
#include <hip/hip_bf16.h>

#define N_NODES 50000
#define IN_F    128
#define HEADS   4
#define HID     64
#define HF      256
#define NEG_SLOPE 0.2f
#define BN_EPS  1e-5f
#define ROW_TILES (N_NODES / 16)   // 3125, exact

typedef __bf16 bf16_t;
typedef bf16_t v8bf  __attribute__((ext_vector_type(8)));
typedef bf16_t v16bf __attribute__((ext_vector_type(16)));
typedef float  v8f   __attribute__((ext_vector_type(8)));

__device__ __forceinline__ v16bf cat8(v8bf lo, v8bf hi) {
    return __builtin_shufflevector(lo, hi, 0, 1, 2, 3, 4, 5, 6, 7,
                                           8, 9, 10, 11, 12, 13, 14, 15);
}

// ---------------------------------------------------------------------------
// f32 -> bf16 streaming convert (2 elements/thread -> v_cvt_pk_bf16_f32)
// ---------------------------------------------------------------------------
__global__ void cvt_bf16_kernel(const float* __restrict__ s, bf16_t* __restrict__ d,
                                long long n2 /* = n/2 */)
{
    long long i = (long long)blockIdx.x * blockDim.x + threadIdx.x;
    if (i >= n2) return;
    float2 v = ((const float2*)s)[i];
    d[2 * i]     = (bf16_t)v.x;
    d[2 * i + 1] = (bf16_t)v.y;
}

// ---------------------------------------------------------------------------
// GEMM1: xw[N,256] = x[N,128] @ W[256,128]^T   (bf16 WMMA, f32 accumulate)
// A/B pre-converted to bf16: inner loop = 4x 16B loads + 1 WMMA.
// block = 256 threads = 8 waves; wave w -> column tile (blockIdx.y*8 + w)
// grid = (3125, 2)
// ---------------------------------------------------------------------------
__global__ __launch_bounds__(256) void gemm1_wmma(
    const bf16_t* __restrict__ xb, const bf16_t* __restrict__ Wb,
    float* __restrict__ xw)
{
    const int wave = threadIdx.x >> 5;
    const int lane = threadIdx.x & 31;
    const int half = lane >> 4;         // 0: lanes 0-15, 1: lanes 16-31
    const int mr   = lane & 15;

    const int rowBase = blockIdx.x * 16;
    const int colBase = (blockIdx.y * 8 + wave) * 16;

    const bf16_t* __restrict__ xrow = xb + (size_t)(rowBase + mr) * IN_F;
    const bf16_t* __restrict__ wrow = Wb + (size_t)(colBase + mr) * IN_F; // B col = W row

    v8f acc = {};
    #pragma unroll
    for (int k0 = 0; k0 < IN_F; k0 += 32) {
        // A layout (16-bit, 16x32): lanes<16 hold K {k0..+7, k0+16..+23}; lanes>=16 +8
        const int abase = k0 + half * 8;
        v8bf a0 = *(const v8bf*)(xrow + abase);
        v8bf a1 = *(const v8bf*)(xrow + abase + 16);
        // B layout (32x16): lanes<16 hold K {k0..+7, k0+16..+23}; lanes>=16 +8
        const int bbase = k0 + half * 8;
        v8bf b0 = *(const v8bf*)(wrow + bbase);
        v8bf b1 = *(const v8bf*)(wrow + bbase + 16);

        acc = __builtin_amdgcn_wmma_f32_16x16x32_bf16(
                  false, cat8(a0, a1), false, cat8(b0, b1), (short)0, acc, false, false);
    }
    #pragma unroll
    for (int r = 0; r < 8; ++r) {
        const int m = rowBase + r + half * 8;     // C layout: VGPR r -> M=r (+8 upper half)
        xw[(size_t)m * HF + colBase + mr] = acc[r];
    }
}

// ---------------------------------------------------------------------------
// alpha_src/dst[n,h] = sum_f xw[n,h*64+f] * att[h,f]
// ---------------------------------------------------------------------------
__global__ void alphas_kernel(const float* __restrict__ xw,
                              const float* __restrict__ a_src,
                              const float* __restrict__ a_dst,
                              float* __restrict__ o_src, float* __restrict__ o_dst)
{
    int idx = blockIdx.x * blockDim.x + threadIdx.x;
    if (idx >= N_NODES * HEADS) return;
    int n = idx >> 2, h = idx & 3;
    const float* row = xw + (size_t)n * HF + h * HID;
    const float* as  = a_src + h * HID;
    const float* ad  = a_dst + h * HID;
    float s1 = 0.f, s2 = 0.f;
    #pragma unroll 8
    for (int f = 0; f < HID; ++f) { float v = row[f]; s1 += v * as[f]; s2 += v * ad[f]; }
    o_src[idx] = s1; o_dst[idx] = s2;
}

__global__ void fill_u32(unsigned* __restrict__ p, unsigned val, long long n)
{
    long long i = (long long)blockIdx.x * blockDim.x + threadIdx.x;
    if (i < n) p[i] = val;
}

__device__ __forceinline__ float lrelu(float v) { return v > 0.f ? v : v * NEG_SLOPE; }

// segment-max via signed-max / unsigned-min bit trick (no CAS loop)
__global__ void edge_max_kernel(const long long* __restrict__ ei, int e_raw, int e_tot,
                                const float* __restrict__ asrc, const float* __restrict__ adst,
                                unsigned* __restrict__ mmax)
{
    int idx = blockIdx.x * blockDim.x + threadIdx.x;
    if (idx >= e_tot * HEADS) return;
    int e = idx >> 2, h = idx & 3;
    long long s, d;
    if (e < e_raw) { s = ei[e]; d = ei[e_raw + e]; } else { s = d = (long long)(e - e_raw); }
    float v = lrelu(asrc[s * 4 + h] + adst[d * 4 + h]);
    unsigned* addr = mmax + d * 4 + h;
    if (v >= 0.f) atomicMax((int*)addr, __float_as_int(v));
    else          atomicMin(addr, (unsigned)__float_as_int(v));
}

__global__ void edge_sum_kernel(const long long* __restrict__ ei, int e_raw, int e_tot,
                                const float* __restrict__ asrc, const float* __restrict__ adst,
                                const float* __restrict__ mmax, float* __restrict__ denom)
{
    int idx = blockIdx.x * blockDim.x + threadIdx.x;
    if (idx >= e_tot * HEADS) return;
    int e = idx >> 2, h = idx & 3;
    long long s, d;
    if (e < e_raw) { s = ei[e]; d = ei[e_raw + e]; } else { s = d = (long long)(e - e_raw); }
    float v = lrelu(asrc[s * 4 + h] + adst[d * 4 + h]);
    atomicAdd(denom + d * 4 + h, __expf(v - mmax[d * 4 + h]));
}

// one warp per edge: lane covers columns {lane, lane+32, ..., lane+224}
__global__ void edge_msg_kernel(const long long* __restrict__ ei, int e_raw, int e_tot,
                                const float* __restrict__ asrc, const float* __restrict__ adst,
                                const float* __restrict__ mmax, const float* __restrict__ denom,
                                const float* __restrict__ xw, float* __restrict__ outg)
{
    int warp = (blockIdx.x * blockDim.x + threadIdx.x) >> 5;
    int lane = threadIdx.x & 31;
    if (warp >= e_tot) return;
    long long s, d;
    if (warp < e_raw) { s = ei[warp]; d = ei[e_raw + warp]; }
    else              { s = d = (long long)(warp - e_raw); }
    float al[HEADS];
    #pragma unroll
    for (int h = 0; h < HEADS; ++h) {
        float v = lrelu(asrc[s * 4 + h] + adst[d * 4 + h]);
        al[h] = __expf(v - mmax[d * 4 + h]) / (denom[d * 4 + h] + 1e-16f);
    }
    const float* srow = xw + (size_t)s * HF;
    float*       drow = outg + (size_t)d * HF;
    #pragma unroll
    for (int j = 0; j < 8; ++j) {
        int c = lane + 32 * j;
        atomicAdd(&drow[c], al[c >> 6] * srow[c]);
    }
}

// fold bias + BN into per-column scale/shift:  bn(v) = v*scale + shift
__global__ void bnprep_kernel(const float* __restrict__ bias, const float* __restrict__ gamma,
                              const float* __restrict__ beta, const float* __restrict__ mean,
                              const float* __restrict__ var,
                              float* __restrict__ scale, float* __restrict__ shift)
{
    int c = threadIdx.x;
    if (c >= HF) return;
    float sc = gamma[c] * rsqrtf(var[c] + BN_EPS);
    scale[c] = sc;
    shift[c] = (bias[c] - mean[c]) * sc + beta[c];
}

// ---------------------------------------------------------------------------
// GEMM2 (fused BN+ReLU on A load): all_emb[N,64] += relu(bn(out_gat)) @ linW^T + linb
// B pre-converted to bf16. block = 128 = 4 waves; wave -> col tile. grid = 3125.
// ---------------------------------------------------------------------------
__global__ __launch_bounds__(128) void gemm2_wmma(
    const float* __restrict__ outg, const float* __restrict__ scale,
    const float* __restrict__ shift, const bf16_t* __restrict__ lWb,
    const float* __restrict__ linb, float* __restrict__ all_emb)
{
    const int wave = threadIdx.x >> 5;
    const int lane = threadIdx.x & 31;
    const int half = lane >> 4;
    const int mr   = lane & 15;

    const int rowBase = blockIdx.x * 16;
    const int colBase = wave * 16;

    const float*  __restrict__ arow = outg + (size_t)(rowBase + mr) * HF;
    const bf16_t* __restrict__ wrow = lWb  + (size_t)(colBase + mr) * HF;

    v8f acc = {};
    for (int k0 = 0; k0 < HF; k0 += 32) {
        v16bf a;
        const int abase = k0 + half * 8;
        #pragma unroll
        for (int j = 0; j < 8; ++j) {
            int k1 = abase + j, k2 = abase + 16 + j;
            float v1 = fmaxf(arow[k1] * scale[k1] + shift[k1], 0.f);
            float v2 = fmaxf(arow[k2] * scale[k2] + shift[k2], 0.f);
            a[j] = (bf16_t)v1; a[8 + j] = (bf16_t)v2;
        }
        const int bbase = k0 + half * 8;
        v8bf b0 = *(const v8bf*)(wrow + bbase);
        v8bf b1 = *(const v8bf*)(wrow + bbase + 16);

        acc = __builtin_amdgcn_wmma_f32_16x16x32_bf16(
                  false, a, false, cat8(b0, b1), (short)0, acc, false, false);
    }
    const int c = colBase + mr;
    const float lb = linb[c];
    #pragma unroll
    for (int r = 0; r < 8; ++r) {
        const int m = rowBase + r + half * 8;
        all_emb[(size_t)m * HID + c] += acc[r] + lb;   // unique (m,c) per lane/r: no atomic
    }
}

// one warp per node: L2 row normalize
__global__ void norm_kernel(const float* __restrict__ all_emb, float* __restrict__ out)
{
    int warp = (blockIdx.x * blockDim.x + threadIdx.x) >> 5;
    int lane = threadIdx.x & 31;
    if (warp >= N_NODES) return;
    const float* row = all_emb + (size_t)warp * HID;
    float v0 = row[lane], v1 = row[lane + 32];
    float ss = v0 * v0 + v1 * v1;
    #pragma unroll
    for (int o = 16; o > 0; o >>= 1) ss += __shfl_xor(ss, o, 32);
    float inv = rsqrtf(ss);
    out[(size_t)warp * HID + lane]      = v0 * inv;
    out[(size_t)warp * HID + lane + 32] = v1 * inv;
}

// ---------------------------------------------------------------------------
extern "C" void kernel_launch(void* const* d_in, const int* in_sizes, int n_in,
                              void* d_out, int out_size, void* d_ws, size_t ws_size,
                              hipStream_t stream)
{
    const float*     sppmi  = (const float*)d_in[0];
    const float*     Wm     = (const float*)d_in[1];
    const float*     attS   = (const float*)d_in[2];
    const float*     attD   = (const float*)d_in[3];
    const float*     bias   = (const float*)d_in[4];
    const float*     gamma  = (const float*)d_in[5];
    const float*     beta   = (const float*)d_in[6];
    const float*     mean   = (const float*)d_in[7];
    const float*     var    = (const float*)d_in[8];
    const float*     linW   = (const float*)d_in[9];
    const float*     linb   = (const float*)d_in[10];
    const long long* ei     = (const long long*)d_in[11];
    float*           out    = (float*)d_out;

    const int e_raw = in_sizes[11] / 2;        // 800000
    const int e_tot = e_raw + N_NODES;         // + self loops

    // workspace carve-up (~132 MB)
    char* ws = (char*)d_ws;
    float*  xw      = (float*)ws;  ws += (size_t)N_NODES * HF * sizeof(float);
    float*  outg    = (float*)ws;  ws += (size_t)N_NODES * HF * sizeof(float);
    float*  all_emb = (float*)ws;  ws += (size_t)N_NODES * HID * sizeof(float);
    bf16_t* xb      = (bf16_t*)ws; ws += (size_t)N_NODES * IN_F * sizeof(bf16_t);
    bf16_t* Wb      = (bf16_t*)ws; ws += (size_t)HF * IN_F * sizeof(bf16_t);
    bf16_t* lWb     = (bf16_t*)ws; ws += (size_t)HID * HF * sizeof(bf16_t);
    float*  asrc    = (float*)ws;  ws += (size_t)N_NODES * HEADS * sizeof(float);
    float*  adst    = (float*)ws;  ws += (size_t)N_NODES * HEADS * sizeof(float);
    float*  mmax    = (float*)ws;  ws += (size_t)N_NODES * HEADS * sizeof(float);
    float*  denom   = (float*)ws;  ws += (size_t)N_NODES * HEADS * sizeof(float);
    float*  bnsc    = (float*)ws;  ws += HF * sizeof(float);
    float*  bnsh    = (float*)ws;  ws += HF * sizeof(float);

    const long long nEmb = (long long)N_NODES * HID;
    const long long nHF  = (long long)N_NODES * HF;
    const long long nNH  = (long long)N_NODES * HEADS;
    const long long nX2  = (long long)N_NODES * IN_F / 2;
    const long long nW2  = (long long)HF * IN_F / 2;
    const long long nL2  = (long long)HID * HF / 2;

    fill_u32<<<(int)((nEmb + 255) / 256), 256, 0, stream>>>((unsigned*)all_emb, 0u, nEmb);

    for (int i = 0; i < 3; ++i) {
        const float* x_i   = sppmi + (size_t)i * N_NODES * IN_F;
        const float* W_i   = Wm    + (size_t)i * HF * IN_F;
        const float* aS_i  = attS  + (size_t)i * HEADS * HID;
        const float* aD_i  = attD  + (size_t)i * HEADS * HID;
        const float* b_i   = bias  + (size_t)i * HF;
        const float* g_i   = gamma + (size_t)i * HF;
        const float* be_i  = beta  + (size_t)i * HF;
        const float* m_i   = mean  + (size_t)i * HF;
        const float* v_i   = var   + (size_t)i * HF;
        const float* lW_i  = linW  + (size_t)i * HID * HF;
        const float* lb_i  = linb  + (size_t)i * HID;

        // 0) stage bf16 copies of GEMM operands (one streaming pass)
        cvt_bf16_kernel<<<(int)((nX2 + 255) / 256), 256, 0, stream>>>(x_i, xb, nX2);
        cvt_bf16_kernel<<<(int)((nW2 + 255) / 256), 256, 0, stream>>>(W_i, Wb, nW2);
        cvt_bf16_kernel<<<(int)((nL2 + 255) / 256), 256, 0, stream>>>(lW_i, lWb, nL2);

        // 1) xw = x @ W^T  (WMMA bf16, K loop fully unrolled)
        gemm1_wmma<<<dim3(ROW_TILES, 2), 256, 0, stream>>>(xb, Wb, xw);

        // 2) attention logits per node/head
        alphas_kernel<<<(N_NODES * HEADS + 255) / 256, 256, 0, stream>>>(
            xw, aS_i, aD_i, asrc, adst);

        // 3) reset segment buffers: mmax = -inf bits, denom = 0, outg = 0
        fill_u32<<<(int)((nNH + 255) / 256), 256, 0, stream>>>((unsigned*)mmax, 0xFF800000u, nNH);
        fill_u32<<<(int)((nNH + 255) / 256), 256, 0, stream>>>((unsigned*)denom, 0u, nNH);
        fill_u32<<<(int)((nHF + 255) / 256), 256, 0, stream>>>((unsigned*)outg, 0u, nHF);

        // 4) segment softmax + message scatter (stays in 192MB L2)
        int tEH = e_tot * HEADS;
        edge_max_kernel<<<(tEH + 255) / 256, 256, 0, stream>>>(ei, e_raw, e_tot, asrc, adst,
                                                               (unsigned*)mmax);
        edge_sum_kernel<<<(tEH + 255) / 256, 256, 0, stream>>>(ei, e_raw, e_tot, asrc, adst,
                                                               mmax, denom);
        edge_msg_kernel<<<(e_tot + 7) / 8, 256, 0, stream>>>(ei, e_raw, e_tot, asrc, adst,
                                                             mmax, denom, xw, outg);

        // 5) fused bias+BN prep, then BN+ReLU+GEMM2 (WMMA) accumulating into all_emb
        bnprep_kernel<<<1, HF, 0, stream>>>(b_i, g_i, be_i, m_i, v_i, bnsc, bnsh);
        gemm2_wmma<<<ROW_TILES, 128, 0, stream>>>(outg, bnsc, bnsh, lWb, lb_i, all_emb);
    }

    // 6) row L2 normalize into d_out
    norm_kernel<<<(N_NODES * 32 + 255) / 256, 256, 0, stream>>>(all_emb, out);
}
